// DDBasic_75763223101533
// MI455X (gfx1250) — matrix operand order repaired and checked by reading, not compiled, not run on hardware
//
#include <hip/hip_runtime.h>

typedef __attribute__((ext_vector_type(16))) _Float16 v16h;
typedef __attribute__((ext_vector_type(8)))  _Float16 v8h;
typedef __attribute__((ext_vector_type(8)))  float    v8f;

#define NP     25
#define WREG   27136      // packed f16 per (branch,patch): (2+26+25)*512
#define L2OFF  1024
#define L3OFF  14336
#define ZBASE  9800       // start of always-zero region (halves)
#define SLAB   10088      // 392 + 3136 + 6272 = 9800, + 288 zero pad (576B reach)
#define WAVES  2

// ---------------- Kernel 1: bicubic upsample 25->50 (Keys a=-0.5, half-pixel) ----
__global__ void k_upsample(const float* __restrict__ x, _Float16* __restrict__ xup,
                           int total)
{
  int t = blockIdx.x * 256 + threadIdx.x;
  if (t >= total) return;
  int gx = t % 50; int r = t / 50; int gy = r % 50; int bc = r / 50;
  const float* src = x + (size_t)bc * 625;
  int my = gy >> 1, py = gy & 1;
  int mx = gx >> 1, px = gx & 1;
  constexpr float W0[4] = {-0.0234375f, 0.2265625f, 0.8671875f, -0.0703125f};
  constexpr float W1[4] = {-0.0703125f, 0.8671875f, 0.2265625f, -0.0234375f};
  float acc = 0.f;
  #pragma unroll
  for (int i = 0; i < 4; ++i) {
    int iy = my - 2 + py + i; iy = iy < 0 ? 0 : (iy > 24 ? 24 : iy);
    float wy = py ? W1[i] : W0[i];
    float ra = 0.f;
    #pragma unroll
    for (int j = 0; j < 4; ++j) {
      int ix = mx - 2 + px + j; ix = ix < 0 ? 0 : (ix > 24 ? 24 : ix);
      float wx = px ? W1[j] : W0[j];
      ra += wx * src[iy * 25 + ix];
    }
    acc += wy * ra;
  }
  xup[t] = (_Float16)acc;
}

// ---------------- Kernel 2: pack weights into WMMA B-tiles (f16) ----------------
// K ordering is tap-major blocks: k -> tap = k/CI, ci = k%CI  (CI = 2/16/32).
// B tile (K=32 x N=16): lane = n + 16*h; 16 halves/lane, klocal = s<8 ? 8h+s : 16+8h+(s-8).
__global__ void k_pack(const float* __restrict__ phiw1, const float* __restrict__ phiw2,
                       const float* __restrict__ phiw3, const float* __restrict__ pw1,
                       const float* __restrict__ pw2,  const float* __restrict__ pw3,
                       _Float16* __restrict__ wpk, int total)
{
  int t = blockIdx.x * 256 + threadIdx.x;
  if (t >= total) return;
  int br  = t / (NP * WREG);
  int rem = t % (NP * WREG);
  int p   = rem / WREG;
  int off = rem % WREG;
  int lane = (off >> 4) & 31;
  int s    = off & 15;
  int n      = lane & 15;
  int klocal = (s < 8) ? (((lane & 16) ? 8 : 0) + s) : (16 + ((lane & 16) ? 8 : 0) + (s - 8));
  int tile   = off >> 9;
  int kstep, nt, CI, CO; const float* w;
  if (off < L2OFF)      { kstep = tile;            nt = 0;        CI = 2;  CO = 16; w = br ? pw1 : phiw1; }
  else if (off < L3OFF) { int tt = tile - 2; kstep = tt >> 1; nt = tt & 1; CI = 16; CO = 32; w = br ? pw2 : phiw2; }
  else                  { kstep = tile - 28;       nt = 0;        CI = 32; CO = 1;  w = br ? pw3 : phiw3; }
  int k   = kstep * 32 + klocal;
  int co  = nt * 16 + n;
  int tap = k / CI, ci = k % CI;
  float v = 0.f;
  if (tap < 25 && co < CO)
    v = w[((size_t)p * CO + co) * (CI * 25) + ci * 25 + tap];
  wpk[t] = (_Float16)v;
}

// ---------------- shared epilogue: bias (+ReLU) -> LDS pixel-major, or global ----
template <int NTILES, int COUT, bool RELU, bool FINAL>
__device__ __forceinline__ void epilogue(const v8f (&acc)[7][NTILES],
                                         _Float16* __restrict__ dst,
                                         const float* __restrict__ bias,
                                         float* __restrict__ gout,
                                         int lane, int pi, int pj)
{
  const int n = lane & 15, khi = lane >> 4;
  #pragma unroll
  for (int nt = 0; nt < NTILES; ++nt) {
    int co = nt * 16 + n;
    float bv = (co < COUT) ? bias[co] : 0.f;
    #pragma unroll
    for (int mt = 0; mt < 7; ++mt) {
      int pixbase = mt * 16 + khi * 8;
      if constexpr (FINAL) {
        if (n == 0) {
          #pragma unroll
          for (int r = 0; r < 8; ++r) {
            int pix = pixbase + r;
            bool ok = (mt < 6) || (pix < 100);      // guard folds away for mt<6
            if (ok) {
              int yy = pix / 10, xx = pix % 10;
              gout[(pi * 10 + yy) * 50 + (pj * 10 + xx)] = acc[mt][nt][r] + bv;
            }
          }
        }
      } else {
        if (co < COUT) {
          #pragma unroll
          for (int r = 0; r < 8; ++r) {
            int pix = pixbase + r;
            bool ok = (mt < 6) || (pix < 100);
            if (ok) {
              float v = acc[mt][nt][r] + bv;
              if (RELU) v = fmaxf(v, 0.f);
              int yy = pix / 10, xx = pix % 10;
              dst[((yy + 2) * 14 + (xx + 2)) * COUT + co] = (_Float16)v;
            }
          }
        }
      }
    }
  }
}

__device__ __forceinline__ int rowb(int pix) {       // clamped pixel -> padded row
  if (pix > 99) pix = 99;
  return (pix / 10) * 14 + (pix % 10);
}

#define WMMA(Amat, Bmat, Cmat) \
  __builtin_amdgcn_wmma_f32_16x16x32_f16(false, (Amat), false, (Bmat), (short)0, (Cmat), false, false)

// ---------------- layer 1: 2->16, K = tap*2+ci (50), b32 tap-chunk loads ---------
__device__ __forceinline__ void conv_l1(const _Float16* __restrict__ src,   // [196][2]
                                        _Float16* __restrict__ dst,         // [196][16]
                                        const _Float16* __restrict__ wtiles,
                                        const float* __restrict__ bias,
                                        int lane, int zoffB)
{
  v8f acc[7][1];
  #pragma unroll
  for (int mt = 0; mt < 7; ++mt) acc[mt][0] = v8f{};
  const int n = lane & 15, khi = lane >> 4;
  int lbB[7];
  #pragma unroll
  for (int mt = 0; mt < 7; ++mt) lbB[mt] = rowb(mt * 16 + n) * 4;
  const char* srcB = (const char*)src;

  #pragma unroll 1
  for (int ks = 0; ks < 2; ++ks) {
    v16h Bt = *(const v16h*)(wtiles + (size_t)ks * 512 + lane * 16);
    const int ksu = __builtin_amdgcn_readfirstlane(ks);
    int offT[16];                      // uniform tap byte-offsets (SALU)
    #pragma unroll
    for (int tt = 0; tt < 16; ++tt) {
      int tap = ksu * 16 + tt;
      int o   = ((tap / 5) * 14 + (tap % 5)) * 4;
      offT[tt] = (tap < 25) ? o : zoffB;
    }
    // A halves 2j,2j+1 <- tap chunk: taps {khi*4+q, 8+khi*4+q}
    int offs[8];
    #pragma unroll
    for (int j = 0; j < 8; ++j) {
      int g = (j < 4) ? 0 : 8;
      int q = j & 3;
      offs[j] = khi ? offT[g + 4 + q] : offT[g + q];
    }
    #pragma unroll
    for (int mt = 0; mt < 7; ++mt) {
      union { v16h v; unsigned u[8]; } au;
      #pragma unroll
      for (int j = 0; j < 8; ++j)
        au.u[j] = *(const unsigned*)(srcB + (lbB[mt] + offs[j]));
      acc[mt][0] = WMMA(au.v, Bt, acc[mt][0]);
    }
  }
  epilogue<1, 16, true, false>(acc, dst, bias, nullptr, lane, 0, 0);
}

// ---------------- layer 2: 16->32, K-step = 2 taps x 16ch, b128 A-loads ----------
__device__ __forceinline__ void conv_l2(const _Float16* __restrict__ src,   // [196][16]
                                        _Float16* __restrict__ dst,         // [196][32]
                                        const _Float16* __restrict__ wtiles,
                                        const float* __restrict__ bias, int lane)
{
  v8f acc[7][2];
  #pragma unroll
  for (int mt = 0; mt < 7; ++mt) { acc[mt][0] = v8f{}; acc[mt][1] = v8f{}; }
  const int n = lane & 15, khi = lane >> 4;
  int lbB[7];
  #pragma unroll
  for (int mt = 0; mt < 7; ++mt) lbB[mt] = rowb(mt * 16 + n) * 32 + khi * 16;
  const char* srcB = (const char*)src;

  #pragma unroll 1
  for (int ks = 0; ks < 12; ++ks) {
    v16h Bt0 = *(const v16h*)(wtiles + (size_t)(ks * 2 + 0) * 512 + lane * 16);
    v16h Bt1 = *(const v16h*)(wtiles + (size_t)(ks * 2 + 1) * 512 + lane * 16);
    const int tap0 = __builtin_amdgcn_readfirstlane(ks * 2);
    const int tap1 = tap0 + 1;
    const int off0 = ((tap0 / 5) * 14 + (tap0 % 5)) * 32;
    const int off1 = ((tap1 / 5) * 14 + (tap1 % 5)) * 32;
    #pragma unroll
    for (int mt = 0; mt < 7; ++mt) {
      v8h lo = *(const v8h*)(srcB + (lbB[mt] + off0));
      v8h hi = *(const v8h*)(srcB + (lbB[mt] + off1));
      v16h A = __builtin_shufflevector(lo, hi, 0, 1, 2, 3, 4, 5, 6, 7,
                                       8, 9, 10, 11, 12, 13, 14, 15);
      acc[mt][0] = WMMA(A, Bt0, acc[mt][0]);
      acc[mt][1] = WMMA(A, Bt1, acc[mt][1]);
    }
  }
  { // peeled tap 24 (upper 16 K rows are zero-padded in B; zero A upper half)
    v16h Bt0 = *(const v16h*)(wtiles + (size_t)24 * 512 + lane * 16);
    v16h Bt1 = *(const v16h*)(wtiles + (size_t)25 * 512 + lane * 16);
    constexpr int off24 = (4 * 14 + 4) * 32;
    v8h z = {};
    #pragma unroll
    for (int mt = 0; mt < 7; ++mt) {
      v8h lo = *(const v8h*)(srcB + (lbB[mt] + off24));
      v16h A = __builtin_shufflevector(lo, z, 0, 1, 2, 3, 4, 5, 6, 7,
                                       8, 9, 10, 11, 12, 13, 14, 15);
      acc[mt][0] = WMMA(A, Bt0, acc[mt][0]);
      acc[mt][1] = WMMA(A, Bt1, acc[mt][1]);
    }
  }
  epilogue<2, 32, true, false>(acc, dst, bias, nullptr, lane, 0, 0);
}

// ---------------- layer 3: 32->1, K-step = 1 tap x 32ch, b128 A-loads ------------
__device__ __forceinline__ void conv_l3(const _Float16* __restrict__ src,   // [196][32]
                                        const _Float16* __restrict__ wtiles,
                                        const float* __restrict__ bias,
                                        float* __restrict__ gout,
                                        int lane, int pi, int pj)
{
  v8f acc[7][1];
  #pragma unroll
  for (int mt = 0; mt < 7; ++mt) acc[mt][0] = v8f{};
  const int n = lane & 15, khi = lane >> 4;
  int lbB[7];
  #pragma unroll
  for (int mt = 0; mt < 7; ++mt) lbB[mt] = rowb(mt * 16 + n) * 64 + khi * 16;
  const char* srcB = (const char*)src;

  #pragma unroll 1
  for (int ks = 0; ks < 25; ++ks) {
    v16h Bt = *(const v16h*)(wtiles + (size_t)ks * 512 + lane * 16);
    const int tap = __builtin_amdgcn_readfirstlane(ks);
    const int off = ((tap / 5) * 14 + (tap % 5)) * 64;
    #pragma unroll
    for (int mt = 0; mt < 7; ++mt) {
      v8h lo = *(const v8h*)(srcB + (lbB[mt] + off));
      v8h hi = *(const v8h*)(srcB + (lbB[mt] + off + 32));
      v16h A = __builtin_shufflevector(lo, hi, 0, 1, 2, 3, 4, 5, 6, 7,
                                       8, 9, 10, 11, 12, 13, 14, 15);
      acc[mt][0] = WMMA(A, Bt, acc[mt][0]);
    }
  }
  epilogue<1, 1, false, true>(acc, nullptr, bias, gout, lane, pi, pj);
}

// ---------------- Kernel 3: fused 3-layer patch CNN, one wave per (b,p,br) -------
__global__ __launch_bounds__(32 * WAVES) void k_fused(
    const _Float16* __restrict__ xup, const _Float16* __restrict__ wpk,
    const float* __restrict__ phib1, const float* __restrict__ phib2,
    const float* __restrict__ phib3, const float* __restrict__ pb1,
    const float* __restrict__ pb2,   const float* __restrict__ pb3,
    float* __restrict__ out)
{
  __shared__ _Float16 smem[WAVES * SLAB];
  const int lane = threadIdx.x & 31;
  const int wv   = threadIdx.x >> 5;
  const int b  = blockIdx.x * WAVES + wv;
  const int p  = blockIdx.y;
  const int br = blockIdx.z;
  const int pi = p / 5, pj = p % 5;

  _Float16* slab = smem + wv * SLAB;
  _Float16* xp = slab;                  // [196][2]  pixel-major, zero halo rows
  _Float16* h1 = slab + 392;            // [196][16]
  _Float16* h2 = slab + 392 + 3136;     // [196][32]
  // slab[ZBASE..SLAB) stays zero: load target for padded K slots (layer 1)

  // zero slab (halo rows + zero region)
  uint4* zv = reinterpret_cast<uint4*>(slab);        // SLAB*2 = 20176B = 1261*16B
  uint4 z; z.x = z.y = z.z = z.w = 0u;
  for (int i = lane; i < 1261; i += 32) zv[i] = z;
  __syncthreads();

  // stage interior of upsampled patch into LDS (pixel-major)
  const _Float16* xg = xup + (size_t)b * 2 * 2500;
  for (int idx = lane; idx < 200; idx += 32) {
    int pix = idx >> 1, c = idx & 1;
    int y = pix / 10, x = pix % 10;
    xp[((y + 2) * 14 + (x + 2)) * 2 + c] =
        xg[c * 2500 + (pi * 10 + y) * 50 + (pj * 10 + x)];
  }
  __syncthreads();

  const _Float16* wb = wpk + (size_t)(br * NP + p) * WREG;
  const float* b1 = (br ? pb1 : phib1) + p * 16;
  const float* b2 = (br ? pb2 : phib2) + p * 32;
  const float* b3 = (br ? pb3 : phib3) + p;

  conv_l1(xp, h1, wb, b1, lane, (ZBASE - 0) * 2);
  __syncthreads();
  conv_l2(h1, h2, wb + L2OFF, b2, lane);
  __syncthreads();
  float* gout = out + ((size_t)b * 2 + br) * 2500;
  conv_l3(h2, wb + L3OFF, b3, gout, lane, pi, pj);
}

// -------------------------------- launch ----------------------------------------
extern "C" void kernel_launch(void* const* d_in, const int* in_sizes, int n_in,
                              void* d_out, int out_size, void* d_ws, size_t ws_size,
                              hipStream_t stream) {
  const float* x      = (const float*)d_in[0];
  const float* phiw1  = (const float*)d_in[1];
  const float* phib1  = (const float*)d_in[2];
  const float* phiw2  = (const float*)d_in[3];
  const float* phib2  = (const float*)d_in[4];
  const float* phiw3  = (const float*)d_in[5];
  const float* phib3  = (const float*)d_in[6];
  const float* pw1    = (const float*)d_in[7];
  const float* pb1    = (const float*)d_in[8];
  const float* pw2    = (const float*)d_in[9];
  const float* pb2    = (const float*)d_in[10];
  const float* pw3    = (const float*)d_in[11];
  const float* pb3    = (const float*)d_in[12];
  float* out = (float*)d_out;

  const int B = in_sizes[0] / (2 * 625);          // 2048
  _Float16* xup = (_Float16*)d_ws;                 // B*2*2500 halves
  _Float16* wpk = xup + (size_t)B * 5000;          // 2*25*27136 halves

  int upN = B * 2 * 2500;
  k_upsample<<<(upN + 255) / 256, 256, 0, stream>>>(x, xup, upN);

  int pkN = 2 * NP * WREG;
  k_pack<<<(pkN + 255) / 256, 256, 0, stream>>>(phiw1, phiw2, phiw3,
                                                pw1, pw2, pw3, wpk, pkN);

  dim3 grid(B / WAVES, NP, 2);
  k_fused<<<grid, 32 * WAVES, 0, stream>>>(xup, wpk, phib1, phib2, phib3,
                                           pb1, pb2, pb3, out);
}
